// Block_34565896798623
// MI455X (gfx1250) — compile-verified, matrix operand order checked by reading, not run on hardware
//
#include <hip/hip_runtime.h>
#include <hip/hip_bf16.h>

#define Bsz 4
#define Tsz 2048
#define Csz 1024
#define Hsz 16

typedef _Float16 h8   __attribute__((ext_vector_type(8)));
typedef _Float16 v16h __attribute__((ext_vector_type(16)));
typedef float    v8f  __attribute__((ext_vector_type(8)));
typedef unsigned int u32x4 __attribute__((ext_vector_type(4)));
typedef int          i32x4 __attribute__((ext_vector_type(4)));
typedef int          i32x8 __attribute__((ext_vector_type(8)));

union F16Frag { v16h v; h8 h[2]; };

// ---------------------------------------------------------------------------
// TDM helper: async-load a 128-row x 64-half 2D tile (row stride = stride
// halfs in global) into LDS at lds_off, with 16B padding after each 128B row
// (effective LDS row stride = 72 halfs). Tracked by TENSORcnt.
// D# encoding per CDNA5 ISA 8.3-8.6. 6-arg builtin (clang-23 / therock HIP).
// ---------------------------------------------------------------------------
__device__ __forceinline__ void tdm_load_tile_128x64(unsigned lds_off,
                                                     const _Float16* gsrc,
                                                     unsigned stride_elems)
{
    unsigned long long ga = (unsigned long long)(uintptr_t)gsrc;
    u32x4 g0;
    g0.x = 1u;                                           // count=1, user mode
    g0.y = lds_off;                                      // lds_addr [63:32]
    g0.z = (unsigned)(ga & 0xffffffffu);                 // global_addr lo
    g0.w = (unsigned)((ga >> 32) & 0x01ffffffu)          // global_addr[56:32]
         | 0x80000000u;                                  // type=2 ("image")
    i32x8 g1;
    g1[0] = (1 << 16)        // data_size = 1 -> 2 bytes
          | (1 << 20)        // pad_enable
          | (4 << 22)        // pad_interval: 4 -> every 32 DWORDs (128B row)
          | (3 << 25);       // pad_amount: 3 -> 4 DWORDs (16B) padding
    g1[1] = (64 << 16);                       // tensor_dim0 = 64 (bits 79:48)
    g1[2] = (128 << 16);                      // tensor_dim1 = 128 (bits 111:80)
    g1[3] = (64 << 16);                       // tile_dim0 = 64 (bits 127:112)
    g1[4] = 128;                              // tile_dim1 = 128, tile_dim2 = 0
    g1[5] = (int)stride_elems;                // tensor_dim0_stride lo32
    g1[6] = 0;                                // stride hi16 | dim1_stride lo16
    g1[7] = 0;
    i32x4 g2 = {0, 0, 0, 0};
    i32x4 g3 = {0, 0, 0, 0};
    i32x8 g4 = {0, 0, 0, 0, 0, 0, 0, 0};
    __builtin_amdgcn_tensor_load_to_lds(g0, g1, g2, g3, g4, 0);
}

// ---------------------------------------------------------------------------
// Transpose + convert weights: WT[n*K + k] = (f16) W[k*N + n]
// ---------------------------------------------------------------------------
__global__ __launch_bounds__(256)
void wconv_kernel(const float* __restrict__ W, _Float16* __restrict__ WT,
                  int K, int N)
{
    __shared__ _Float16 t[32][40];
    int tx = threadIdx.x & 31, ty = threadIdx.x >> 5;   // ty: 0..7
    int k0 = blockIdx.x * 32, n0 = blockIdx.y * 32;
    #pragma unroll
    for (int i = 0; i < 4; ++i) {
        int k = k0 + ty * 4 + i;
        t[tx][ty * 4 + i] = (_Float16)W[(size_t)k * N + n0 + tx];
    }
    __syncthreads();
    #pragma unroll
    for (int i = 0; i < 4; ++i) {
        int n = n0 + ty * 4 + i;
        WT[(size_t)n * K + k0 + tx] = t[ty * 4 + i][tx];
    }
}

// ---------------------------------------------------------------------------
// LayerNorm: fp32 input row [1024] -> f16 normalized output
// ---------------------------------------------------------------------------
__global__ __launch_bounds__(256)
void layernorm_kernel(const float* __restrict__ x, const float* __restrict__ g,
                      const float* __restrict__ bb, _Float16* __restrict__ out)
{
    int row = blockIdx.x;
    int tid = threadIdx.x;
    const float4* xr = (const float4*)(x + (size_t)row * Csz);
    float4 v = xr[tid];

    __shared__ float red[8];
    __shared__ float stat[2];

    float s = v.x + v.y + v.z + v.w;
    #pragma unroll
    for (int m = 16; m; m >>= 1) s += __shfl_xor(s, m, 32);
    if ((tid & 31) == 0) red[tid >> 5] = s;
    __syncthreads();
    if (tid == 0) {
        float t = 0.f;
        #pragma unroll
        for (int i = 0; i < 8; ++i) t += red[i];
        stat[0] = t * (1.0f / Csz);
    }
    __syncthreads();
    float mu = stat[0];
    float dx = v.x - mu, dy = v.y - mu, dz = v.z - mu, dw = v.w - mu;
    float q = dx*dx + dy*dy + dz*dz + dw*dw;
    #pragma unroll
    for (int m = 16; m; m >>= 1) q += __shfl_xor(q, m, 32);
    __syncthreads();
    if ((tid & 31) == 0) red[tid >> 5] = q;
    __syncthreads();
    if (tid == 0) {
        float t = 0.f;
        #pragma unroll
        for (int i = 0; i < 8; ++i) t += red[i];
        stat[1] = rsqrtf(t * (1.0f / Csz) + 1e-5f);
    }
    __syncthreads();
    float rs = stat[1];
    int c = tid * 4;
    _Float16* o = out + (size_t)row * Csz + c;
    o[0] = (_Float16)(dx * rs * g[c+0] + bb[c+0]);
    o[1] = (_Float16)(dy * rs * g[c+1] + bb[c+1]);
    o[2] = (_Float16)(dz * rs * g[c+2] + bb[c+2]);
    o[3] = (_Float16)(dw * rs * g[c+3] + bb[c+3]);
}

// ---------------------------------------------------------------------------
// GEMM: Out[M,N] = f(A16[M,K] @ WT16[N,K]^T + bias) (+ residual fp32)
// 128x128x64 block tile, double-buffered TDM staging, 8 wave32s (4x2),
// each wave computes 32x64 via 2x4 WMMA accumulators, 16 WMMAs / k-stage.
// ---------------------------------------------------------------------------
#define LDT 72   // LDS row stride in halfs (64 data + 8 pad, set by TDM pad)

template<int GELU, int RESID, int OUTF16>
__global__ __launch_bounds__(256)
void gemm_kernel(const _Float16* __restrict__ A, const _Float16* __restrict__ WT,
                 const float* __restrict__ bias, const float* resid,
                 _Float16* __restrict__ out16, float* out32,
                 int M, int N, int K)
{
    __shared__ _Float16 As[2][128 * LDT];
    __shared__ _Float16 Bs[2][128 * LDT];

    int tid = threadIdx.x;
    int bm = blockIdx.x * 128, bn = blockIdx.y * 128;
    int wv = tid >> 5, lane = tid & 31;
    int wr = wv >> 1, wc = wv & 1;
    int lrow = lane & 15, lhalf = lane >> 4;

    unsigned asOff[2], bsOff[2];
    asOff[0] = (unsigned)(uintptr_t)(void*)&As[0][0];
    asOff[1] = (unsigned)(uintptr_t)(void*)&As[1][0];
    bsOff[0] = (unsigned)(uintptr_t)(void*)&Bs[0][0];
    bsOff[1] = (unsigned)(uintptr_t)(void*)&Bs[1][0];

    const _Float16* aBase = A  + (size_t)bm * K;
    const _Float16* bBase = WT + (size_t)bn * K;

    v8f acc[2][4] = {};

    int nk = K >> 6;
    if (wv == 0) {
        tdm_load_tile_128x64(asOff[0], aBase, (unsigned)K);
        tdm_load_tile_128x64(bsOff[0], bBase, (unsigned)K);
    }

    for (int i = 0; i < nk; ++i) {
        int cur = i & 1;
        if (wv == 0) {
            if (i + 1 < nk) {
                tdm_load_tile_128x64(asOff[cur ^ 1], aBase + (i + 1) * 64, (unsigned)K);
                tdm_load_tile_128x64(bsOff[cur ^ 1], bBase + (i + 1) * 64, (unsigned)K);
                __builtin_amdgcn_s_wait_tensorcnt(2);   // stage `cur` done, next pair in flight
            } else {
                __builtin_amdgcn_s_wait_tensorcnt(0);   // last stage: drain
            }
        }
        __syncthreads();   // stage `cur` visible to all waves

        const _Float16* as = As[cur];
        const _Float16* bs = Bs[cur];
        F16Frag afr[2][2], bfr[4][2];
        #pragma unroll
        for (int mt = 0; mt < 2; ++mt)
            #pragma unroll
            for (int kt = 0; kt < 2; ++kt) {
                const _Float16* p = &as[(wr*32 + mt*16 + lrow) * LDT + kt*32 + lhalf*8];
                afr[mt][kt].h[0] = *(const h8*)p;
                afr[mt][kt].h[1] = *(const h8*)(p + 16);
            }
        #pragma unroll
        for (int nt = 0; nt < 4; ++nt)
            #pragma unroll
            for (int kt = 0; kt < 2; ++kt) {
                const _Float16* p = &bs[(wc*64 + nt*16 + lrow) * LDT + kt*32 + lhalf*8];
                bfr[nt][kt].h[0] = *(const h8*)p;
                bfr[nt][kt].h[1] = *(const h8*)(p + 16);
            }
        #pragma unroll
        for (int kt = 0; kt < 2; ++kt)
            #pragma unroll
            for (int mt = 0; mt < 2; ++mt)
                #pragma unroll
                for (int nt = 0; nt < 4; ++nt)
                    acc[mt][nt] = __builtin_amdgcn_wmma_f32_16x16x32_f16(
                        false, afr[mt][kt].v, false, bfr[nt][kt].v,
                        (short)0, acc[mt][nt], false, false);
        __syncthreads();   // everyone done with stage `cur` before TDM reuses it
    }

    // Epilogue
    #pragma unroll
    for (int mt = 0; mt < 2; ++mt) {
        #pragma unroll
        for (int nt = 0; nt < 4; ++nt) {
            int gn = bn + wc*64 + nt*16 + lrow;
            float bv = bias[gn];
            #pragma unroll
            for (int r = 0; r < 8; ++r) {
                int gm = bm + wr*32 + mt*16 + r + lhalf*8;
                float v = acc[mt][nt][r] + bv;
                if (GELU) {
                    float x3 = v * v * v;
                    v = 0.5f * v * (1.0f + tanhf(0.7978845608028654f * (v + 0.044715f * x3)));
                }
                size_t idx = (size_t)gm * N + gn;
                if (OUTF16) out16[idx] = (_Float16)v;
                else        out32[idx] = (RESID ? resid[idx] : 0.0f) + v;
            }
        }
    }
}

// ---------------------------------------------------------------------------
// Flash attention (causal): one block per (b, h, 128-row q-block)
// 8 waves, each wave owns a 16-row q strip; 64-key blocks.
// ---------------------------------------------------------------------------
__global__ __launch_bounds__(256)
void attn_kernel(const _Float16* __restrict__ qkv, _Float16* __restrict__ out)
{
    __shared__ _Float16 Kt[64 * 72];   // [n][k]  (K rows directly = B^T layout)
    __shared__ _Float16 Vt[64 * 72];   // [d][kk] (V transposed)
    __shared__ _Float16 Qs[128 * 72];  // Q stage, reused as per-wave P buffer

    const int LDQ = 3 * Csz;
    int qb = blockIdx.x, bh = blockIdx.y;
    int b = bh >> 4, h = bh & 15;
    size_t rowbase = (size_t)b * Tsz;
    int qoff = h * 64, koff = Csz + h * 64, voff = 2 * Csz + h * 64;

    int tid = threadIdx.x, wv = tid >> 5, lane = tid & 31;
    int lrow = lane & 15, lhalf = lane >> 4;

    // Stage Q block (128 x 64 f16)
    {
        int r = tid >> 1, cs = (tid & 1) * 32;
        const _Float16* src = qkv + (rowbase + qb*128 + r) * LDQ + qoff + cs;
        *(h8*)&Qs[r*72 + cs]      = *(const h8*)(src);
        *(h8*)&Qs[r*72 + cs + 8]  = *(const h8*)(src + 8);
        *(h8*)&Qs[r*72 + cs + 16] = *(const h8*)(src + 16);
        *(h8*)&Qs[r*72 + cs + 24] = *(const h8*)(src + 24);
    }
    __syncthreads();
    F16Frag qfr[2];
    #pragma unroll
    for (int kt = 0; kt < 2; ++kt) {
        const _Float16* p = &Qs[(wv*16 + lrow) * 72 + kt*32 + lhalf*8];
        qfr[kt].h[0] = *(const h8*)p;
        qfr[kt].h[1] = *(const h8*)(p + 16);
    }
    __syncthreads();   // Qs now reusable as P buffer

    _Float16* Pw = &Qs[wv * 16 * 72];  // wave-private P strip [16][64]

    float mrow[8], lsum[8];
    v8f oacc[4] = {};
    #pragma unroll
    for (int r = 0; r < 8; ++r) { mrow[r] = -1e30f; lsum[r] = 0.f; }

    int nkb = 2 * qb + 2;
    for (int kb = 0; kb < nkb; ++kb) {
        // Stage K tile [64 keys][64 dims]
        {
            int r = tid >> 2, cs = (tid & 3) * 16;
            const _Float16* src = qkv + (rowbase + kb*64 + r) * LDQ + koff + cs;
            *(h8*)&Kt[r*72 + cs]     = *(const h8*)(src);
            *(h8*)&Kt[r*72 + cs + 8] = *(const h8*)(src + 8);
        }
        // Stage V transposed: Vt[d][kk]
        {
            int kk = tid >> 2, ds = (tid & 3) * 16;
            const _Float16* src = qkv + (rowbase + kb*64 + kk) * LDQ + voff + ds;
            h8 a = *(const h8*)(src), c = *(const h8*)(src + 8);
            #pragma unroll
            for (int j = 0; j < 8; ++j) Vt[(ds + j) * 72 + kk] = a[j];
            #pragma unroll
            for (int j = 0; j < 8; ++j) Vt[(ds + 8 + j) * 72 + kk] = c[j];
        }
        __syncthreads();

        // Scores: S = Q (16x64) * K^T -> 4 tiles of 16x16, 2 WMMAs each
        float sv[4][8];
        #pragma unroll
        for (int nt = 0; nt < 4; ++nt) {
            v8f sacc = {};
            #pragma unroll
            for (int kt = 0; kt < 2; ++kt) {
                F16Frag bfr;
                const _Float16* p = &Kt[(nt*16 + lrow) * 72 + kt*32 + lhalf*8];
                bfr.h[0] = *(const h8*)p;
                bfr.h[1] = *(const h8*)(p + 16);
                sacc = __builtin_amdgcn_wmma_f32_16x16x32_f16(
                    false, qfr[kt].v, false, bfr.v, (short)0, sacc, false, false);
            }
            int kg = kb*64 + nt*16 + lrow;
            #pragma unroll
            for (int r = 0; r < 8; ++r) {
                int qg = qb*128 + wv*16 + r + lhalf*8;
                sv[nt][r] = (kg > qg) ? -1e30f : sacc[r] * 0.125f;
            }
        }

        // Online softmax update (row reductions across 16-lane halves)
        float pnew[4][8];
        #pragma unroll
        for (int r = 0; r < 8; ++r) {
            float rm = fmaxf(fmaxf(sv[0][r], sv[1][r]), fmaxf(sv[2][r], sv[3][r]));
            #pragma unroll
            for (int m = 1; m < 16; m <<= 1) rm = fmaxf(rm, __shfl_xor(rm, m, 32));
            float mnew = fmaxf(mrow[r], rm);
            float scale = __expf(mrow[r] - mnew);
            float rs = 0.f;
            #pragma unroll
            for (int nt = 0; nt < 4; ++nt) {
                float p_ = __expf(sv[nt][r] - mnew);
                pnew[nt][r] = p_;
                rs += p_;
            }
            #pragma unroll
            for (int m = 1; m < 16; m <<= 1) rs += __shfl_xor(rs, m, 32);
            lsum[r] = lsum[r] * scale + rs;
            mrow[r] = mnew;
            #pragma unroll
            for (int dt = 0; dt < 4; ++dt) oacc[dt][r] *= scale;
        }

        // Re-layout P (D-layout -> A-fragment) via wave-private LDS
        #pragma unroll
        for (int nt = 0; nt < 4; ++nt)
            #pragma unroll
            for (int r = 0; r < 8; ++r)
                Pw[(r + lhalf*8) * 72 + nt*16 + lrow] = (_Float16)pnew[nt][r];

        // PV: out += P (16x64) * V (64x64)
        #pragma unroll
        for (int kt = 0; kt < 2; ++kt) {
            F16Frag afr;
            const _Float16* p = &Pw[lrow * 72 + kt*32 + lhalf*8];
            afr.h[0] = *(const h8*)p;
            afr.h[1] = *(const h8*)(p + 16);
            #pragma unroll
            for (int dt = 0; dt < 4; ++dt) {
                F16Frag bfr;
                const _Float16* q = &Vt[(dt*16 + lrow) * 72 + kt*32 + lhalf*8];
                bfr.h[0] = *(const h8*)q;
                bfr.h[1] = *(const h8*)(q + 16);
                oacc[dt] = __builtin_amdgcn_wmma_f32_16x16x32_f16(
                    false, afr.v, false, bfr.v, (short)0, oacc[dt], false, false);
            }
        }
        __syncthreads();
    }

    // Epilogue: out[b*T + q][h*64 + d] = oacc / lsum
    #pragma unroll
    for (int dt = 0; dt < 4; ++dt) {
        int col = h*64 + dt*16 + lrow;
        #pragma unroll
        for (int r = 0; r < 8; ++r) {
            int qrow = qb*128 + wv*16 + r + lhalf*8;
            out[(rowbase + qrow) * Csz + col] = (_Float16)(oacc[dt][r] / lsum[r]);
        }
    }
}

// ---------------------------------------------------------------------------
// Host launch
// ---------------------------------------------------------------------------
extern "C" void kernel_launch(void* const* d_in, const int* in_sizes, int n_in,
                              void* d_out, int out_size, void* d_ws, size_t ws_size,
                              hipStream_t stream)
{
    (void)in_sizes; (void)n_in; (void)out_size; (void)ws_size;

    const float* x       = (const float*)d_in[0];
    const float* ln1_g   = (const float*)d_in[1];
    const float* ln1_b   = (const float*)d_in[2];
    const float* w_attn  = (const float*)d_in[3];
    const float* b_attn  = (const float*)d_in[4];
    const float* w_proj  = (const float*)d_in[5];
    const float* b_proj  = (const float*)d_in[6];
    const float* ln2_g   = (const float*)d_in[7];
    const float* ln2_b   = (const float*)d_in[8];
    const float* w_fc    = (const float*)d_in[9];
    const float* b_fc    = (const float*)d_in[10];
    const float* w_mproj = (const float*)d_in[11];
    const float* b_mproj = (const float*)d_in[12];
    float* out = (float*)d_out;

    char* ws = (char*)d_ws;
    const size_t MB = 1u << 20;
    _Float16* h16     = (_Float16*)(ws);                 // 16 MB [8192,1024]
    _Float16* qkv16   = (_Float16*)(ws + 16*MB);         // 48 MB [8192,3072]
    _Float16* attn16  = (_Float16*)(ws + 64*MB);         // 16 MB [8192,1024]
    _Float16* fc16    = (_Float16*)(ws + 16*MB);         // 64 MB [8192,4096] (reuses dead qkv/attn region)
    _Float16* w_attnT = (_Float16*)(ws + 80*MB);         // 6 MB  [3072,1024]
    _Float16* w_projT = (_Float16*)(ws + 88*MB);         // 2 MB  [1024,1024]
    _Float16* w_fcT   = (_Float16*)(ws + 92*MB);         // 8 MB  [4096,1024]
    _Float16* w_mprojT= (_Float16*)(ws + 100*MB);        // 8 MB  [1024,4096]

    const int M = Bsz * Tsz;  // 8192

    // One-pass weight convert+transpose (fp32 [K,N] -> f16 [N,K])
    wconv_kernel<<<dim3(1024/32, 3072/32), 256, 0, stream>>>(w_attn,  w_attnT,  1024, 3072);
    wconv_kernel<<<dim3(1024/32, 1024/32), 256, 0, stream>>>(w_proj,  w_projT,  1024, 1024);
    wconv_kernel<<<dim3(1024/32, 4096/32), 256, 0, stream>>>(w_fc,    w_fcT,    1024, 4096);
    wconv_kernel<<<dim3(4096/32, 1024/32), 256, 0, stream>>>(w_mproj, w_mprojT, 4096, 1024);

    // LN1 -> h16
    layernorm_kernel<<<M, 256, 0, stream>>>(x, ln1_g, ln1_b, h16);
    // qkv = h16 @ w_attn + b_attn (f16 out)
    gemm_kernel<0,0,1><<<dim3(M/128, 3072/128), 256, 0, stream>>>(
        h16, w_attnT, b_attn, nullptr, qkv16, nullptr, M, 3*Csz, Csz);
    // attention -> attn16
    attn_kernel<<<dim3(Tsz/128, Bsz*Hsz), 256, 0, stream>>>(qkv16, attn16);
    // x1 = x + attn16 @ w_proj + b_proj  (fp32 -> d_out)
    gemm_kernel<0,1,0><<<dim3(M/128, 1024/128), 256, 0, stream>>>(
        attn16, w_projT, b_proj, x, nullptr, out, M, Csz, Csz);
    // LN2 -> h16 (reuse)
    layernorm_kernel<<<M, 256, 0, stream>>>(out, ln2_g, ln2_b, h16);
    // fc16 = gelu(h16 @ w_fc + b_fc) (f16 out)
    gemm_kernel<1,0,1><<<dim3(M/128, 4096/128), 256, 0, stream>>>(
        h16, w_fcT, b_fc, nullptr, fc16, nullptr, M, 4*Csz, Csz);
    // out = x1 + fc16 @ w_mproj + b_mproj (fp32, read+write d_out per element)
    gemm_kernel<0,1,0><<<dim3(M/128, 1024/128), 256, 0, stream>>>(
        fc16, w_mprojT, b_mproj, out, nullptr, out, M, Csz, 4*Csz);
}
